// MoV_75600014344300
// MI455X (gfx1250) — matrix operand order repaired and checked by reading, not compiled, not run on hardware
//
#include <hip/hip_runtime.h>
#include <hip/hip_bf16.h>

typedef __attribute__((ext_vector_type(16))) _Float16 v16h;
typedef __attribute__((ext_vector_type(8)))  _Float16 v8h;
typedef __attribute__((ext_vector_type(8)))  float    v8f;

#define B_   8
#define C_   512
#define H_   96
#define W_   96
#define NH   8
#define HD   64
#define AMP  3
#define KS   7
#define TILE 16
#define KP   22              // TILE + 2*AMP
#define NPIX (TILE * TILE)   // 256
#define KPIX (KP * KP)       // 484
#define SROW 24              // padded score-row length (>=22)
#define PLANE (H_ * W_)      // 9216

// dynamic LDS layout:
//   lds_q : NPIX*HD f16  (32768 B)
//   lds_k : KPIX*HD f16  (61952 B)
//   lds_s : 8 waves * 16 * SROW f32 (12288 B)
#define SMEM_BYTES (NPIX * HD * 2 + KPIX * HD * 2 + 8 * 16 * SROW * 4)

__global__ __launch_bounds__(256)
void mov_wmma_kernel(const float* __restrict__ xq,
                     const float* __restrict__ xk,
                     float* __restrict__ out) {
  extern __shared__ char smem[];
  _Float16* lds_q = (_Float16*)smem;              // [pix][d]
  _Float16* lds_k = lds_q + NPIX * HD;            // [kpix][d]
  float*    lds_s = (float*)(lds_k + KPIX * HD);  // per-wave [16][SROW]

  const int t       = threadIdx.x;
  const int lane    = t & 31;
  const int wv      = t >> 5;
  const int lane_lo = lane & 15;
  const int lane_hi = lane >> 4;

  const int gid = blockIdx.x;      // 288 blocks: (b, th, tw)
  const int b   = gid / 36;
  const int tt  = gid % 36;
  const int h0  = (tt / 6) * TILE;
  const int w0  = (tt % 6) * TILE;

  const int r_own = t >> 4;        // this thread's pixel within the tile
  const int m_own = t & 15;
  const int h_own = h0 + r_own;
  const int w_own = w0 + m_own;

  float* Sw = lds_s + wv * (16 * SROW);

  float acc_di = 0.0f, acc_dj = 0.0f;

  for (int n = 0; n < NH; ++n) {
    __syncthreads();  // previous head's compute done before overwriting LDS

    // ---- stage Q tile (scaled by hd^-0.5 = 1/8) as f16 ----
    {
      const int qbase = ((b * C_ + n) * H_ + h_own) * W_ + w_own;
      #pragma unroll 4
      for (int d = 0; d < HD; ++d) {
        float f = xq[qbase + d * (NH * PLANE)];
        lds_q[t * HD + d] = (_Float16)(f * 0.125f);
      }
    }
    // ---- stage 22x22 K halo patch as f16 (zero OOB) ----
    for (int p = t; p < KPIX; p += 256) {
      const int pr = p / KP, pc = p % KP;
      const int gh = h0 + pr - AMP, gw = w0 + pc - AMP;
      const bool ok = ((unsigned)gh < (unsigned)H_) && ((unsigned)gw < (unsigned)W_);
      const int kbase = ((b * C_ + n) * H_ + gh) * W_ + gw;
      #pragma unroll 4
      for (int d = 0; d < HD; ++d) {
        float f = ok ? xk[kbase + d * (NH * PLANE)] : 0.0f;
        lds_k[p * HD + d] = (_Float16)f;
      }
    }
    __syncthreads();

    float band[49];  // this thread's 49 logits (fully register-resident)

    #pragma unroll
    for (int rr = 0; rr < 2; ++rr) {
      const int r = 2 * wv + rr;   // query row handled by this wave

      // A fragments for K=0..31 and K=32..63 per ISA 16-bit A layout:
      // lanes 0-15 hold K base+{0..7,16..23}; lanes 16-31 hold K base+{8..15,24..31}
      v16h A[2];
      #pragma unroll
      for (int s = 0; s < 2; ++s) {
        const _Float16* ap = lds_q + (r * 16 + lane_lo) * HD + s * 32 + lane_hi * 8;
        v8h alo = *(const v8h*)ap;         // K = s*32 + lane_hi*8 + (0..7)
        v8h ahi = *(const v8h*)(ap + 16);  // K = s*32 + 16 + lane_hi*8 + (0..7)
        #pragma unroll
        for (int e = 0; e < 8; ++e) { A[s][e] = alo[e]; A[s][8 + e] = ahi[e]; }
      }

      #pragma unroll
      for (int kr = 0; kr < KS; ++kr) {
        #pragma unroll
        for (int cs = 0; cs < 2; ++cs) {
          // B fragment: N = lane_lo -> key column (cs*6 + lane_lo) of patch row r+kr,
          // K table order: lanes 0-15 K {0..7,16..23}, lanes 16-31 K {8..15,24..31} (per 32-split)
          const int kcol = lane_lo + cs * 6;
          const int kpix = (r + kr) * KP + kcol;
          v8f D = {};
          #pragma unroll
          for (int s = 0; s < 2; ++s) {
            const _Float16* bp = lds_k + kpix * HD + s * 32 + lane_hi * 8;
            const v8h blo = *(const v8h*)bp;
            const v8h bhi = *(const v8h*)(bp + 16);
            v16h Bf;
            #pragma unroll
            for (int e = 0; e < 8; ++e) { Bf[e] = blo[e]; Bf[8 + e] = bhi[e]; }
            D = __builtin_amdgcn_wmma_f32_16x16x32_f16(false, A[s], false, Bf,
                                                       (short)0, D, false, false);
          }
          // scatter needed score columns to per-wave LDS scratch:
          // cs0 covers key cols 0..15, cs1 contributes cols 16..21 (N>=10)
          #pragma unroll
          for (int g = 0; g < 8; ++g) {
            const int mrow = g + 8 * lane_hi;
            if (cs == 0)            Sw[mrow * SROW + lane_lo]     = D[g];
            else if (lane_lo >= 10) Sw[mrow * SROW + lane_lo + 6] = D[g];
          }
        }
        // band extraction: query m needs key cols m..m+6 (cross-lane via LDS).
        // LDS ops within a wave are in-order, so read-after-write is safe.
        const bool own = (lane_hi == rr);
        #pragma unroll
        for (int j = 0; j < KS; ++j) {
          const float v = Sw[lane_lo * SROW + lane_lo + j];
          if (own) band[kr * KS + j] = v;
        }
      }
    }

    // ---- masked softmax over 49 offsets + offset-weighted sums ----
    float mx = -1.0e30f;
    #pragma unroll
    for (int o = 0; o < 49; ++o) {
      const int di = o / KS - AMP, dj = o % KS - AMP;
      const bool ok = ((unsigned)(h_own + di) < (unsigned)H_) &&
                      ((unsigned)(w_own + dj) < (unsigned)W_);
      const float v = ok ? band[o] : -1000.0f;
      band[o] = v;
      mx = fmaxf(mx, v);
    }
    float sum = 0.0f, sdi = 0.0f, sdj = 0.0f;
    #pragma unroll
    for (int o = 0; o < 49; ++o) {
      const float p = __expf(band[o] - mx);
      sum += p;
      sdi += p * (float)(o / KS - AMP);
      sdj += p * (float)(o % KS - AMP);
    }
    const float inv = 1.0f / sum;
    acc_di += sdi * inv;
    acc_dj += sdj * inv;
  }

  // mean over heads; output layout (B, 2, H, W)
  out[((b * 2 + 0) * H_ + h_own) * W_ + w_own] = acc_di * 0.125f;
  out[((b * 2 + 1) * H_ + h_own) * W_ + w_own] = acc_dj * 0.125f;
}

extern "C" void kernel_launch(void* const* d_in, const int* in_sizes, int n_in,
                              void* d_out, int out_size, void* d_ws, size_t ws_size,
                              hipStream_t stream) {
  (void)in_sizes; (void)n_in; (void)d_ws; (void)ws_size; (void)out_size;
  const float* xq = (const float*)d_in[0];
  const float* xk = (const float*)d_in[1];
  float* out = (float*)d_out;

  // >64KB dynamic LDS: CDNA5 WGP has 320KB — opt in explicitly.
  (void)hipFuncSetAttribute((const void*)mov_wmma_kernel,
                            hipFuncAttributeMaxDynamicSharedMemorySize,
                            SMEM_BYTES);

  const int blocks = B_ * (H_ / TILE) * (W_ / TILE);  // 288
  mov_wmma_kernel<<<blocks, 256, SMEM_BYTES, stream>>>(xq, xk, out);
}
